// BaseMultiheadAttention_31052613550416
// MI455X (gfx1250) — matrix operand order repaired and checked by reading, not compiled, not run on hardware
//
#include <hip/hip_runtime.h>

// dims: B*V=4, S=2048, E=512, H=8, D=64 ; M = 8192 rows
typedef __attribute__((ext_vector_type(16))) __bf16 bf16x16;
typedef __attribute__((ext_vector_type(8)))  float  f32x8;

union AFrag {           // 32 bytes: one WMMA 16-bit A or B fragment (per lane)
    bf16x16 v;
    uint4   q[2];
};

__device__ inline f32x8 zero8() {
    f32x8 z;
#pragma unroll
    for (int i = 0; i < 8; ++i) z[i] = 0.f;
    return z;
}

// -------- CDNA5 async global->LDS copy (ASYNCcnt path), with safe fallback ----
#if defined(__gfx1250__) && __has_builtin(__builtin_amdgcn_global_load_async_to_lds_b128)
#define ASYNC_LDS 1
typedef int v4i_t __attribute__((vector_size(16)));
typedef __attribute__((address_space(1))) v4i_t g_v4i;   // global int4
typedef __attribute__((address_space(3))) v4i_t l_v4i;   // LDS int4
#else
#define ASYNC_LDS 0
#endif

__device__ inline void async_cp16(const void* g, void* l) {
#if ASYNC_LDS
    __builtin_amdgcn_global_load_async_to_lds_b128((g_v4i*)g, (l_v4i*)l, 0, 0);
#else
    *(uint4*)l = *(const uint4*)g;
#endif
}

__device__ inline void async_wait0() {
#if ASYNC_LDS
    asm volatile("s_wait_asynccnt 0x0" ::: "memory");
#endif
}

// ---------------------------------------------------------------- fp32 -> bf16
__global__ __launch_bounds__(256) void cvt_kernel(const float* __restrict__ in,
                                                  __bf16* __restrict__ out, int n) {
    int t = blockIdx.x * 256 + threadIdx.x;
    if (t < n) out[t] = (__bf16)in[t];
}

// ---------------------------------------------------------------- GEMM (bf16 WMMA)
// C[M,N] = A[M,512] * Bw[N,512]^T + bias[N]
// Double-buffered LDS tiles filled by async global->LDS copies.
// MODE 0: QKV projection -> scatter bf16 into Qb/Kb (bvh,s,d) and Vt (bvh,d,s)
// MODE 1: output projection -> fp32 row-major Out[M,N]
template <int MODE>
__global__ __launch_bounds__(256) void gemm_bf16(const __bf16* __restrict__ A,
                                                 const __bf16* __restrict__ Bw,
                                                 const float* __restrict__ bias,
                                                 __bf16* __restrict__ Qb,
                                                 __bf16* __restrict__ Kb,
                                                 __bf16* __restrict__ Vt,
                                                 float* __restrict__ Out, int N) {
    __shared__ __align__(16) __bf16 As[2][128 * 64];
    __shared__ __align__(16) __bf16 Bs[2][128 * 64];

    const int tid  = threadIdx.x;
    const int lane = tid & 31;
    const int wave = tid >> 5;
    const int wm   = wave & 1;      // 2 wave-rows  (64 M each)
    const int wn   = wave >> 1;     // 4 wave-cols  (32 N each)
    const int l15  = lane & 15;
    const int half = lane >> 4;
    const int m0   = blockIdx.x * 128;
    const int n0   = blockIdx.y * 128;

    f32x8 acc[4][2];
#pragma unroll
    for (int mt = 0; mt < 4; ++mt)
#pragma unroll
        for (int nt = 0; nt < 2; ++nt) acc[mt][nt] = zero8();

    // tile = 128x64 bf16 = 1024 x 16B chunks; 256 threads * 4 chunks each
    const int ci0 = tid * 4;
    auto issue = [&](int k0, int buf) {
#pragma unroll
        for (int c = 0; c < 4; ++c) {
            const int ci  = ci0 + c;
            const int row = ci >> 2;
            const int e   = (ci & 3) * 8;
            async_cp16(A  + (size_t)(m0 + row) * 512 + k0 + e, &As[buf][row * 64 + e]);
            async_cp16(Bw + (size_t)(n0 + row) * 512 + k0 + e, &Bs[buf][row * 64 + e]);
        }
    };

    issue(0, 0);
    for (int it = 0; it < 8; ++it) {
        const int cur = it & 1;
        async_wait0();          // this wave's copies into buf 'cur' complete
        __syncthreads();        // all waves' copies visible; prev reads of cur^1 done
        if (it < 7) issue((it + 1) * 64, cur ^ 1);

#pragma unroll
        for (int ks = 0; ks < 2; ++ks) {
            AFrag af[4], bf[2];
#pragma unroll
            for (int mt = 0; mt < 4; ++mt) {
                const int row = wm * 64 + mt * 16 + l15;
                af[mt].q[0] = *(const uint4*)&As[cur][row * 64 + ks * 32 + half * 8];
                af[mt].q[1] = *(const uint4*)&As[cur][row * 64 + ks * 32 + 16 + half * 8];
            }
#pragma unroll
            for (int nt = 0; nt < 2; ++nt) {
                const int col = wn * 32 + nt * 16 + l15;
                bf[nt].q[0] = *(const uint4*)&Bs[cur][col * 64 + ks * 32 + half * 8];
                bf[nt].q[1] = *(const uint4*)&Bs[cur][col * 64 + ks * 32 + 16 + half * 8];
            }
#pragma unroll
            for (int mt = 0; mt < 4; ++mt)
#pragma unroll
                for (int nt = 0; nt < 2; ++nt)
                    acc[mt][nt] = __builtin_amdgcn_wmma_f32_16x16x32_bf16(
                        false, af[mt].v, false, bf[nt].v, (short)0, acc[mt][nt],
                        false, false);
        }
    }

    // epilogue: C layout => element r, lane l: M = r + (l>>4)*8, N = l&15
#pragma unroll
    for (int mt = 0; mt < 4; ++mt)
#pragma unroll
        for (int nt = 0; nt < 2; ++nt)
#pragma unroll
            for (int r = 0; r < 8; ++r) {
                const int m = m0 + wm * 64 + mt * 16 + r + half * 8;
                const int n = n0 + wn * 32 + nt * 16 + l15;
                const float c = acc[mt][nt][r] + bias[n];
                if constexpr (MODE == 0) {
                    // feature order: n = qkv*512 + d*8 + h
                    const int qkv = n >> 9, rr = n & 511;
                    const int d = rr >> 3, h = rr & 7;
                    const int bv = m >> 11, s = m & 2047;
                    const int bvh = bv * 8 + h;
                    const __bf16 hv = (__bf16)c;
                    if (qkv == 0)       Qb[((size_t)bvh * 2048 + s) * 64 + d] = hv;
                    else if (qkv == 1)  Kb[((size_t)bvh * 2048 + s) * 64 + d] = hv;
                    else                Vt[((size_t)bvh * 64 + d) * 2048 + s] = hv;
                } else {
                    Out[(size_t)m * N + n] = c;
                }
            }
}

// ---------------------------------------------------------------- interleaved RoPE (in place)
__global__ __launch_bounds__(256) void rope_kernel(__bf16* __restrict__ Qb,
                                                   __bf16* __restrict__ Kb) {
    const int t   = blockIdx.x * 256 + threadIdx.x;  // 32*2048*32 threads
    const int i   = t & 31;                          // pair index within head_dim
    const int s   = (t >> 5) & 2047;
    const int bvh = t >> 16;
    const float inv = __powf(10000.f, -(float)i * (1.f / 32.f));
    float sn, cs;
    __sincosf((float)s * inv, &sn, &cs);
    const size_t off = ((size_t)bvh * 2048 + s) * 64 + 2 * i;
    {
        float x0 = (float)Qb[off], x1 = (float)Qb[off + 1];
        Qb[off]     = (__bf16)(x0 * cs - x1 * sn);
        Qb[off + 1] = (__bf16)(x1 * cs + x0 * sn);
    }
    {
        float x0 = (float)Kb[off], x1 = (float)Kb[off + 1];
        Kb[off]     = (__bf16)(x0 * cs - x1 * sn);
        Kb[off + 1] = (__bf16)(x1 * cs + x0 * sn);
    }
}

// ---------------------------------------------------------------- flash attention
// grid: (qblock 0..31, bvh 0..31), block = 128 (4 waves, 16 query rows each)
// K/V 64x64 tiles double-buffered in LDS via async copies (4x traffic cut vs
// per-wave global fragment loads); next block prefetched during compute.
__global__ __launch_bounds__(128) void attn_kernel(const __bf16* __restrict__ Qb,
                                                   const __bf16* __restrict__ Kb,
                                                   const __bf16* __restrict__ Vt,
                                                   __bf16* __restrict__ Ob) {
    __shared__ __align__(16) __bf16 Ks[2][64 * 64];
    __shared__ __align__(16) __bf16 Vs[2][64 * 64];
    __shared__ __align__(16) __bf16 Pl[4 * 16 * 64];  // per-wave P tiles

    const int tid  = threadIdx.x;
    const int lane = tid & 31;
    const int wave = tid >> 5;
    const int l15  = lane & 15;
    const int half = lane >> 4;
    const int qblock = 31 - blockIdx.x;   // heavy (long) blocks launch first
    const int bvh    = blockIdx.y;
    const int qrow0  = qblock * 64 + wave * 16;

    const __bf16* Qh = Qb + (size_t)bvh * 2048 * 64;
    const __bf16* Kh = Kb + (size_t)bvh * 2048 * 64;
    const __bf16* Vh = Vt + (size_t)bvh * 64 * 2048;

    // K tile (64x64, contiguous in global) + V tile (64 d-rows of 128B): 512
    // 16B-chunks each; 128 threads * 4 chunks
    const int ci0 = tid * 4;
    auto issueKV = [&](int j, int buf) {
#pragma unroll
        for (int c = 0; c < 4; ++c) {
            const int ci = ci0 + c;
            async_cp16(Kh + (size_t)j * 4096 + ci * 8, &Ks[buf][ci * 8]);
            const int d = ci >> 3, c8 = ci & 7;
            async_cp16(Vh + (size_t)d * 2048 + j * 64 + c8 * 8,
                       &Vs[buf][d * 64 + c8 * 8]);
        }
    };

    // Q fragments for both head-dim k-steps, pre-scaled by 1/sqrt(D)=0.125 (exact in bf16)
    AFrag qa[2];
    const __bf16* qrow = Qh + (size_t)(qrow0 + l15) * 64;
#pragma unroll
    for (int ks = 0; ks < 2; ++ks) {
        qa[ks].q[0] = *(const uint4*)(qrow + ks * 32 + half * 8);
        qa[ks].q[1] = *(const uint4*)(qrow + ks * 32 + 16 + half * 8);
#pragma unroll
        for (int e = 0; e < 16; ++e)
            qa[ks].v[e] = (__bf16)((float)qa[ks].v[e] * 0.125f);
    }

    f32x8 o[4];
#pragma unroll
    for (int nt = 0; nt < 4; ++nt) o[nt] = zero8();
    float mrow[8], lrow[8];
#pragma unroll
    for (int r = 0; r < 8; ++r) { mrow[r] = -1e30f; lrow[r] = 0.f; }

    issueKV(0, 0);
    for (int j = 0; j <= qblock; ++j) {
        const int cur = j & 1;
        async_wait0();        // own copies into buf 'cur' done
        __syncthreads();      // tiles visible; prev iter's reads of cur^1 done
        if (j < qblock) issueKV(j + 1, cur ^ 1);   // prefetch next block

        // ---- S = (Q*scale) @ K^T  (64 keys, from LDS)
        f32x8 sc[4];
#pragma unroll
        for (int nt = 0; nt < 4; ++nt) sc[nt] = zero8();
#pragma unroll
        for (int ks = 0; ks < 2; ++ks)
#pragma unroll
            for (int nt = 0; nt < 4; ++nt) {
                AFrag kb;
                const __bf16* krow =
                    &Ks[cur][(nt * 16 + l15) * 64 + ks * 32 + half * 8];
                kb.q[0] = *(const uint4*)krow;
                kb.q[1] = *(const uint4*)(krow + 16);
                sc[nt] = __builtin_amdgcn_wmma_f32_16x16x32_bf16(
                    false, qa[ks].v, false, kb.v, (short)0, sc[nt], false, false);
            }
        // ---- causal mask on diagonal block
        if (j == qblock) {
#pragma unroll
            for (int nt = 0; nt < 4; ++nt)
#pragma unroll
                for (int r = 0; r < 8; ++r) {
                    const int key = j * 64 + nt * 16 + l15;
                    const int q   = qrow0 + r + half * 8;
                    if (key > q) sc[nt][r] = -1e30f;
                }
        }
        // ---- online softmax (rows live in elements; reduce across 16-lane group)
        float vmax[8];
#pragma unroll
        for (int r = 0; r < 8; ++r)
            vmax[r] = fmaxf(fmaxf(sc[0][r], sc[1][r]), fmaxf(sc[2][r], sc[3][r]));
#pragma unroll
        for (int off = 8; off >= 1; off >>= 1)
#pragma unroll
            for (int r = 0; r < 8; ++r)
                vmax[r] = fmaxf(vmax[r], __shfl_xor(vmax[r], off, 16));
        float alpha[8], rsum[8];
#pragma unroll
        for (int r = 0; r < 8; ++r) {
            const float mn = fmaxf(mrow[r], vmax[r]);
            alpha[r] = __expf(mrow[r] - mn);
            mrow[r]  = mn;
            rsum[r]  = 0.f;
        }
#pragma unroll
        for (int nt = 0; nt < 4; ++nt)
#pragma unroll
            for (int r = 0; r < 8; ++r) {
                const float p = __expf(sc[nt][r] - mrow[r]);
                rsum[r] += p;
                Pl[(wave * 16 + r + half * 8) * 64 + nt * 16 + l15] = (__bf16)p;
            }
#pragma unroll
        for (int off = 8; off >= 1; off >>= 1)
#pragma unroll
            for (int r = 0; r < 8; ++r) rsum[r] += __shfl_xor(rsum[r], off, 16);
#pragma unroll
        for (int r = 0; r < 8; ++r) lrow[r] = lrow[r] * alpha[r] + rsum[r];
#pragma unroll
        for (int nt = 0; nt < 4; ++nt)
#pragma unroll
            for (int r = 0; r < 8; ++r) o[nt][r] *= alpha[r];

        __syncthreads();   // P (C-layout) -> LDS complete
        // ---- O += P @ V (V from LDS)
#pragma unroll
        for (int ks2 = 0; ks2 < 2; ++ks2) {
            AFrag pa;
            const __bf16* prow = Pl + (wave * 16 + l15) * 64 + ks2 * 32;
            pa.q[0] = *(const uint4*)(prow + half * 8);
            pa.q[1] = *(const uint4*)(prow + 16 + half * 8);
#pragma unroll
            for (int nt = 0; nt < 4; ++nt) {
                AFrag vb;
                const __bf16* vcol =
                    &Vs[cur][(nt * 16 + l15) * 64 + ks2 * 32 + half * 8];
                vb.q[0] = *(const uint4*)vcol;
                vb.q[1] = *(const uint4*)(vcol + 16);
                o[nt] = __builtin_amdgcn_wmma_f32_16x16x32_bf16(
                    false, pa.v, false, vb.v, (short)0, o[nt], false, false);
            }
        }
    }

    // ---- normalize + store bf16 into (bv, s, h*64+d) for the WO GEMM
    const int bv = bvh >> 3, h = bvh & 7;
#pragma unroll
    for (int nt = 0; nt < 4; ++nt)
#pragma unroll
        for (int r = 0; r < 8; ++r) {
            const int q = qrow0 + r + half * 8;
            const int d = nt * 16 + l15;
            const float val = o[nt][r] / lrow[r];
            Ob[((size_t)bv * 2048 + q) * 512 + h * 64 + d] = (__bf16)val;
        }
}

// ---------------------------------------------------------------- launcher
extern "C" void kernel_launch(void* const* d_in, const int* in_sizes, int n_in,
                              void* d_out, int out_size, void* d_ws, size_t ws_size,
                              hipStream_t stream) {
    const float* x      = (const float*)d_in[1];
    const float* wqkv_w = (const float*)d_in[2];
    const float* wqkv_b = (const float*)d_in[3];
    const float* wo_w   = (const float*)d_in[4];
    const float* wo_b   = (const float*)d_in[5];
    float* out = (float*)d_out;

    char* ws = (char*)d_ws;
    __bf16* Xb    = (__bf16*)(ws + 0);          //  8 MB  X bf16 [8192][512]
    __bf16* Wqkvb = (__bf16*)(ws + 8388608);    //  1.5MB wqkv bf16 [1536][512]
    __bf16* Wob   = (__bf16*)(ws + 9961472);    //  0.5MB wo bf16 [512][512]
    __bf16* Qb    = (__bf16*)(ws + 10485760);   //  8 MB  Q bf16 (bvh,s,d)
    __bf16* Kb    = (__bf16*)(ws + 18874368);   //  8 MB  K bf16 (bvh,s,d)
    __bf16* Vt    = (__bf16*)(ws + 27262976);   //  8 MB  V bf16 (bvh,d,s)
    __bf16* Ab    = (__bf16*)(ws + 35651584);   //  8 MB  attn out bf16 [8192][512]

    cvt_kernel<<<16384, 256, 0, stream>>>(x, Xb, 8192 * 512);
    cvt_kernel<<<3072, 256, 0, stream>>>(wqkv_w, Wqkvb, 1536 * 512);
    cvt_kernel<<<1024, 256, 0, stream>>>(wo_w, Wob, 512 * 512);

    gemm_bf16<0><<<dim3(64, 12), 256, 0, stream>>>(Xb, Wqkvb, wqkv_b,
                                                   Qb, Kb, Vt, nullptr, 1536);
    rope_kernel<<<8192, 256, 0, stream>>>(Qb, Kb);
    attn_kernel<<<dim3(32, 32), 128, 0, stream>>>(Qb, Kb, Vt, Ab);
    gemm_bf16<1><<<dim3(64, 4), 256, 0, stream>>>(Ab, Wob, wo_b,
                                                  nullptr, nullptr, nullptr, out, 512);
}
